// SudokuNN_13889924235660
// MI455X (gfx1250) — matrix operand order repaired and checked by reading, not compiled, not run on hardware
//
#include <hip/hip_runtime.h>
#include <hip/hip_bf16.h>

typedef __attribute__((ext_vector_type(16))) _Float16 v16h;
typedef __attribute__((ext_vector_type(8)))  float    v8f;
typedef __attribute__((ext_vector_type(4)))  int      v4i;

#define NNODES 10368
#define NTILES 648          // NNODES / 16
#define NEDGES 207360
#define TSTEPS 8
#define HDIM   84
#define HP     96           // padded hidden (3 k-chunks, 6 n-tiles)

#if defined(__HIP_DEVICE_COMPILE__) && __has_builtin(__builtin_amdgcn_global_load_async_to_lds_b128)
#define HAVE_ASYNC_LDS 1
#endif

// ---------------------------------------------------------------------------
// Async global -> LDS staging (16-byte granules). Falls back to a synchronous
// 16B register copy if the builtin is unavailable on this toolchain.
// Builtin signature (from clang diagnostic): (v4i AS1*, v4i AS3*, Ii, Ii).
// ---------------------------------------------------------------------------
__device__ __forceinline__ void async_cp16(const _Float16* g, _Float16* l) {
#ifdef HAVE_ASYNC_LDS
  __builtin_amdgcn_global_load_async_to_lds_b128(
      (__attribute__((address_space(1))) v4i*)(g),
      (__attribute__((address_space(3))) v4i*)(l), 0, 0);
#else
  *(uint4*)l = *(const uint4*)g;
#endif
}
__device__ __forceinline__ void async_wait() {
#ifdef HAVE_ASYNC_LDS
  asm volatile("s_wait_asynccnt 0x0" ::: "memory");
#endif
}
__device__ __forceinline__ void lds_read_fence() {   // prior ds reads done before async overwrite
#ifdef HAVE_ASYNC_LDS
  asm volatile("s_wait_dscnt 0x0" ::: "memory");
#endif
}

// ---------------------------------------------------------------------------
// WMMA fragment helpers (CDNA5 16x16x32 f16, wave32)
// A 16x32 f16: lane L -> M=L&15, K-base=(L>>4)*8 ; elems 0..7 -> K+i, 8..15 -> K+16+i
// B 32x16 f16 from W stored [O x Kp] row-major: lane L -> col N=L&15, K-base=(L>>4)*8 ; elems 0..7 -> K+i, 8..15 -> K+16+i
// C/D f32:    lane L -> N=L&15, rows (L>>4)*8 + v
// ---------------------------------------------------------------------------
__device__ __forceinline__ v16h load_A_frag(const _Float16* p, int ld) {
  int lane = threadIdx.x & 31;
  const _Float16* row = p + (lane & 15) * ld + ((lane >> 4) << 3);
  v16h a;
#pragma unroll
  for (int i = 0; i < 8; ++i) a[i] = row[i];
#pragma unroll
  for (int i = 0; i < 8; ++i) a[8 + i] = row[16 + i];
  return a;
}

__device__ __forceinline__ v16h load_B_frag(const _Float16* p, int ld) {
  int lane = threadIdx.x & 31;
  const _Float16* col = p + (lane & 15) * ld + ((lane >> 4) << 3);
  v16h b;
#pragma unroll
  for (int i = 0; i < 16; ++i) b[i] = col[i + ((i >> 3) << 3)];
  return b;
}

__device__ __forceinline__ v8f wmma_f16(v16h a, v16h b, v8f c) {
  return __builtin_amdgcn_wmma_f32_16x16x32_f16(false, a, false, b, (short)0, c,
                                                false, false);
}

// One 16-row x (OT*16)-col layer. A-fragments hoisted: loaded once, reused
// across all OT output tiles. OT loop unrolled x2 so the scheduler can clause
// the next tile's B loads under the current tile's WMMAs.
template <int KC, int OT>
__device__ __forceinline__ void gemm_tile16(const _Float16* in, int ldin,
                                            const _Float16* W, int Kp,
                                            const float* bias, int biasN,
                                            bool relu, _Float16* outH, int ldo) {
  int lane = threadIdx.x & 31;
  int n = lane & 15;
  int mb = (lane >> 4) << 3;
  v16h afr[KC];
#pragma unroll
  for (int kc = 0; kc < KC; ++kc) afr[kc] = load_A_frag(in + kc * 32, ldin);
#pragma unroll 2
  for (int ot = 0; ot < OT; ++ot) {
    v8f acc = {};
#pragma unroll
    for (int kc = 0; kc < KC; ++kc) {
      v16h bf = load_B_frag(W + (ot * 16) * Kp + kc * 32, Kp);
      acc = wmma_f16(afr[kc], bf, acc);
    }
    int ch = ot * 16 + n;
    float bv = (ch < biasN) ? bias[ch] : 0.f;
#pragma unroll
    for (int v = 0; v < 8; ++v) {
      float val = acc[v] + bv;
      if (relu) val = fmaxf(val, 0.f);
      outH[(mb + v) * ldo + ch] = (_Float16)val;
    }
  }
}

__device__ __forceinline__ float sigmoidf_(float x) { return 1.f / (1.f + expf(-x)); }

// ---------------------------------------------------------------------------
// Init / weight-prep kernels (run every launch: graph-replay safe)
// ---------------------------------------------------------------------------
__global__ void zero_f32_kernel(float* p, int n) {
  for (int i = blockIdx.x * blockDim.x + threadIdx.x; i < n; i += gridDim.x * blockDim.x)
    p[i] = 0.f;
}
__global__ void zero_f16_kernel(_Float16* p, int n) {
  for (int i = blockIdx.x * blockDim.x + threadIdx.x; i < n; i += gridDim.x * blockDim.x)
    p[i] = (_Float16)0.f;
}

// generic zero-pad fp32 -> f16  (src [sr x sc] -> dst [dr x dc])
__global__ void pad_weight_kernel(const float* __restrict__ src, _Float16* __restrict__ dst,
                                  int sr, int sc, int dr, int dc) {
  int i = blockIdx.x * blockDim.x + threadIdx.x;
  if (i >= dr * dc) return;
  int r = i / dc, c = i % dc;
  float v = (r < sr && c < sc) ? src[r * sc + c] : 0.f;
  dst[i] = (_Float16)v;
}

// msg_W0 [84 x 168] -> [96 x 192]: cols 0..83 <- src 0..83, 96..179 <- src 84..167
__global__ void pad_msgw0_kernel(const float* __restrict__ src, _Float16* __restrict__ dst) {
  int i = blockIdx.x * blockDim.x + threadIdx.x;
  if (i >= 96 * 192) return;
  int r = i / 192, c = i % 192;
  float v = 0.f;
  if (r < 84) {
    if (c < 84)                  v = src[r * 168 + c];
    else if (c >= 96 && c < 180) v = src[r * 168 + 84 + (c - 96)];
  }
  dst[i] = (_Float16)v;
}

// [Wih(336x168) | Whh(336x84)] -> [336 x 288] matching A = [x(96)|m(96)|h(96)]
__global__ void pack_lstm_kernel(const float* __restrict__ wih, const float* __restrict__ whh,
                                 _Float16* __restrict__ dst) {
  int i = blockIdx.x * blockDim.x + threadIdx.x;
  if (i >= 336 * 288) return;
  int r = i / 288, c = i % 288;
  float v = 0.f;
  if (c < 84)                   v = wih[r * 168 + c];
  else if (c >= 96 && c < 180)  v = wih[r * 168 + 84 + (c - 96)];
  else if (c >= 192 && c < 276) v = whh[r * 84 + (c - 192)];
  dst[i] = (_Float16)v;
}

// ---------------------------------------------------------------------------
// Input MLP: x = mlp4(emb[q]); also seeds hmsg = x.  4 waves/block, 1 tile/wave.
// ---------------------------------------------------------------------------
__global__ void __launch_bounds__(128) input_mlp_kernel(
    const int* __restrict__ q, const float* __restrict__ emb,
    const _Float16* __restrict__ W0, const _Float16* __restrict__ W1,
    const _Float16* __restrict__ W2, const _Float16* __restrict__ W3,
    const float* __restrict__ b0, const float* __restrict__ b1,
    const float* __restrict__ b2, const float* __restrict__ b3,
    _Float16* __restrict__ x_out, _Float16* __restrict__ hmsg) {
  __shared__ __align__(16) _Float16 ldsA[4][16 * 32];
  __shared__ __align__(16) _Float16 ldsP[4][16 * HP];
  __shared__ __align__(16) _Float16 ldsQ[4][16 * HP];
  int wave = threadIdx.x >> 5, lane = threadIdx.x & 31;
  int tile = blockIdx.x * 4 + wave;
  if (tile >= NTILES) return;
  int nb = tile * 16;

  __builtin_prefetch(W1, 0, 2);
  __builtin_prefetch(W2, 0, 2);
  __builtin_prefetch(W3, 0, 2);

  _Float16* A = ldsA[wave];
  for (int idx = lane; idx < 16 * 32; idx += 32) {
    int r = idx >> 5, c = idx & 31;
    float v = (c < 16) ? emb[q[nb + r] * 16 + c] : 0.f;
    A[idx] = (_Float16)v;
  }
  gemm_tile16<1, 6>(A, 32, W0, 32, b0, HDIM, true, ldsP[wave], HP);
  gemm_tile16<3, 6>(ldsP[wave], HP, W1, HP, b1, HDIM, true,  ldsQ[wave], HP);
  gemm_tile16<3, 6>(ldsQ[wave], HP, W2, HP, b2, HDIM, true,  ldsP[wave], HP);
  gemm_tile16<3, 6>(ldsP[wave], HP, W3, HP, b3, HDIM, false, ldsQ[wave], HP);

  for (int idx = lane; idx < 16 * HP; idx += 32) {
    _Float16 v = ldsQ[wave][idx];
    int r = idx / HP, c = idx % HP;
    x_out[(nb + r) * HP + c] = v;
    hmsg[(nb + r) * HP + c] = v;
  }
}

// ---------------------------------------------------------------------------
// Edge MLP + segment sum. Block owns 16 dst nodes = 320 contiguous edges
// (dst[e] == e/20 by construction). 4 waves x 5 edge-tiles of 16 edges.
// Staging is async DMA straight into LDS (no VGPR round trip).
// ---------------------------------------------------------------------------
__global__ void __launch_bounds__(128) msg_kernel(
    const _Float16* __restrict__ h, const int* __restrict__ srcIdx,
    const _Float16* __restrict__ W0, const _Float16* __restrict__ W1,
    const _Float16* __restrict__ W2, const _Float16* __restrict__ W3,
    const float* __restrict__ b0, const float* __restrict__ b1,
    const float* __restrict__ b2, const float* __restrict__ b3,
    _Float16* __restrict__ m_out) {
  __shared__ __align__(16) _Float16 ldsA[4][16 * 192];   // 24 KB
  __shared__ __align__(16) _Float16 ldsP[4][16 * HP];    // 12 KB
  __shared__ __align__(16) _Float16 ldsQ[4][16 * HP];    // 12 KB
  __shared__ __align__(16) float    accum[16 * HP];      //  6 KB
  int wave = threadIdx.x >> 5, lane = threadIdx.x & 31;
  int dstbase = blockIdx.x * 16;

  __builtin_prefetch(W0, 0, 2);
  __builtin_prefetch(W1, 0, 2);
  __builtin_prefetch(W2, 0, 2);
  __builtin_prefetch(W3, 0, 2);

  for (int i = threadIdx.x; i < 16 * HP; i += 128) accum[i] = 0.f;
  __syncthreads();

  for (int t = wave; t < 20; t += 4) {
    int ebase = dstbase * 20 + t * 16;
    // ---- stage 16 edges x 192 halves = 384 16B-granules, async to LDS ----
    lds_read_fence();
#pragma unroll 1
    for (int it = 0; it < 12; ++it) {
      int chunk = it * 32 + lane;          // 0..383
      int r = chunk / 24;                  // edge row in tile
      int c = chunk % 24;                  // 16B granule: 0-11 = h[src], 12-23 = h[dst]
      int e = ebase + r;
      int node = (c < 12) ? srcIdx[e] : (e / 20);
      int cc = (c < 12) ? c : (c - 12);
      async_cp16(h + node * HP + cc * 8, ldsA[wave] + r * 192 + c * 8);
    }
    async_wait();

    gemm_tile16<6, 6>(ldsA[wave], 192, W0, 192, b0, HDIM, true, ldsP[wave], HP);
    gemm_tile16<3, 6>(ldsP[wave], HP, W1, HP, b1, HDIM, true,  ldsQ[wave], HP);
    gemm_tile16<3, 6>(ldsQ[wave], HP, W2, HP, b2, HDIM, true,  ldsP[wave], HP);

    // layer 4: accumulate (with bias) straight into per-dst sums via LDS atomics
    int n = lane & 15, mb = (lane >> 4) << 3;
    v16h afr[3];
#pragma unroll
    for (int kc = 0; kc < 3; ++kc) afr[kc] = load_A_frag(ldsP[wave] + kc * 32, HP);
#pragma unroll 2
    for (int ot = 0; ot < 6; ++ot) {
      v8f acc = {};
#pragma unroll
      for (int kc = 0; kc < 3; ++kc) {
        v16h bf = load_B_frag(W3 + (ot * 16) * HP + kc * 32, HP);
        acc = wmma_f16(afr[kc], bf, acc);
      }
      int ch = ot * 16 + n;
      if (ch < HDIM) {
        float bv = b3[ch];
#pragma unroll
        for (int v = 0; v < 8; ++v) {
          int r = mb + v;
          int dl = (t * 16 + r) / 20;          // local dst 0..15
          atomicAdd(&accum[dl * HP + ch], acc[v] + bv);
        }
      }
    }
  }
  __syncthreads();
  for (int i = threadIdx.x; i < 16 * HP; i += 128) {
    int c = i % HP;
    m_out[dstbase * HP + i] = (_Float16)((c < HDIM) ? accum[i] : 0.f);
  }
}

// ---------------------------------------------------------------------------
// LSTM cell: gates = [x|m|h_rec] (K=288) x Wc[336 x 288]; 1 wave / 16 nodes.
// ---------------------------------------------------------------------------
__global__ void __launch_bounds__(32) lstm_kernel(
    const _Float16* __restrict__ x, const _Float16* __restrict__ m,
    _Float16* hrec,  // read (recurrent in) + write (new h)
    const _Float16* __restrict__ Wc, float* __restrict__ c_state,
    _Float16* __restrict__ hmsg) {
  __shared__ __align__(16) _Float16 A[16 * 288];       //  9 KB
  __shared__ __align__(16) float    gates[16 * 336];   // 21 KB
  int lane = threadIdx.x;
  int nb = blockIdx.x * 16;

  __builtin_prefetch(Wc, 0, 2);

  // stage A = [x(96)|m(96)|h(96)] per row: 16 rows x 36 granules, async
#pragma unroll 1
  for (int it = 0; it < 18; ++it) {
    int chunk = it * 32 + lane;              // 0..575
    int r = chunk / 36;
    int c = chunk % 36;
    int nd = nb + r;
    const _Float16* g = (c < 12) ? (x + nd * HP + c * 8)
                      : (c < 24) ? (m + nd * HP + (c - 12) * 8)
                                 : (hrec + nd * HP + (c - 24) * 8);
    async_cp16(g, A + r * 288 + c * 8);
  }
  async_wait();

  int n = lane & 15, mb = (lane >> 4) << 3;
  v16h afr[9];
#pragma unroll
  for (int kc = 0; kc < 9; ++kc) afr[kc] = load_A_frag(A + kc * 32, 288);
#pragma unroll 2
  for (int ot = 0; ot < 21; ++ot) {          // 336 gate channels = 21 tiles exactly
    v8f acc = {};
#pragma unroll
    for (int kc = 0; kc < 9; ++kc) {
      v16h bf = load_B_frag(Wc + (ot * 16) * 288 + kc * 32, 288);
      acc = wmma_f16(afr[kc], bf, acc);
    }
#pragma unroll
    for (int v = 0; v < 8; ++v) gates[(mb + v) * 336 + ot * 16 + n] = acc[v];
  }
  __syncthreads();

  for (int idx = lane; idx < 16 * HDIM; idx += 32) {
    int r = idx / HDIM, ch = idx % HDIM;
    int nd = nb + r;
    float gi = gates[r * 336 + ch];
    float gf = gates[r * 336 + 84 + ch];
    float gg = gates[r * 336 + 168 + ch];
    float go = gates[r * 336 + 252 + ch];
    float cold = c_state[nd * HDIM + ch];
    float cn = sigmoidf_(gf) * cold + sigmoidf_(gi) * tanhf(gg);
    float hn = sigmoidf_(go) * tanhf(cn);
    c_state[nd * HDIM + ch] = cn;
    _Float16 hh = (_Float16)hn;
    hrec[nd * HP + ch] = hh;   // pad cols stay 0 from init
    hmsg[nd * HP + ch] = hh;
  }
}

// ---------------------------------------------------------------------------
// Readout: logits = h @ outW.T + b ; argmax + log-softmax loss contribution.
// ---------------------------------------------------------------------------
__global__ void __launch_bounds__(32) out_kernel(
    const _Float16* __restrict__ h, const _Float16* __restrict__ Wo,
    const float* __restrict__ bo, const int* __restrict__ lab,
    float* __restrict__ preds, float* __restrict__ lossAcc, int t) {
  __shared__ __align__(16) _Float16 A[16 * HP];
  __shared__ __align__(16) float    L[16 * 16];
  int lane = threadIdx.x;
  int nb = blockIdx.x * 16;

  // stage 16 contiguous padded rows (3 KB) async to LDS
#pragma unroll 1
  for (int it = 0; it < 6; ++it) {
    int chunk = it * 32 + lane;              // 0..191
    async_cp16(h + nb * HP + chunk * 8, A + chunk * 8);
  }
  async_wait();

  v16h afr[3];
#pragma unroll
  for (int kc = 0; kc < 3; ++kc) afr[kc] = load_A_frag(A + kc * 32, HP);
  v8f acc = {};
#pragma unroll
  for (int kc = 0; kc < 3; ++kc) {
    v16h bf = load_B_frag(Wo + kc * 32, HP);
    acc = wmma_f16(afr[kc], bf, acc);
  }
  int n = lane & 15, mb = (lane >> 4) << 3;
  float bv = (n < 9) ? bo[n] : 0.f;
#pragma unroll
  for (int v = 0; v < 8; ++v) L[(mb + v) * 16 + n] = acc[v] + bv;
  __syncthreads();

  for (int r = lane; r < 16; r += 32) {   // lanes 0..15
    float mx = L[r * 16 + 0];
    int am = 0;
    for (int j = 1; j < 9; ++j) {
      float v = L[r * 16 + j];
      if (v > mx) { mx = v; am = j; }
    }
    float se = 0.f;
    for (int j = 0; j < 9; ++j) se += expf(L[r * 16 + j] - mx);
    float lse = mx + logf(se);
    int nd = nb + r;
    float lp = L[r * 16 + lab[nd]] - lse;
    preds[t * NNODES + nd] = (float)am;
    atomicAdd(lossAcc, lp);
  }
}

__global__ void finalize_kernel(const float* __restrict__ lossAcc, float* __restrict__ out) {
  out[TSTEPS * NNODES] = -lossAcc[0] / (float)(TSTEPS * NNODES);
}

// ---------------------------------------------------------------------------
extern "C" void kernel_launch(void* const* d_in, const int* in_sizes, int n_in,
                              void* d_out, int out_size, void* d_ws, size_t ws_size,
                              hipStream_t stream) {
  (void)in_sizes; (void)n_in; (void)out_size; (void)ws_size;
  const int*   q       = (const int*)d_in[0];
  const int*   lab     = (const int*)d_in[1];
  const int*   src     = (const int*)d_in[2];
  const float* emb     = (const float*)d_in[4];
  const float* lstmWih = (const float*)d_in[5];
  const float* lstmWhh = (const float*)d_in[6];
  const float* outW    = (const float*)d_in[7];
  const float* outb    = (const float*)d_in[8];
  const float* inW[4]  = {(const float*)d_in[9],  (const float*)d_in[11],
                          (const float*)d_in[13], (const float*)d_in[15]};
  const float* inB[4]  = {(const float*)d_in[10], (const float*)d_in[12],
                          (const float*)d_in[14], (const float*)d_in[16]};
  const float* msgW[4] = {(const float*)d_in[17], (const float*)d_in[19],
                          (const float*)d_in[21], (const float*)d_in[23]};
  const float* msgB[4] = {(const float*)d_in[18], (const float*)d_in[20],
                          (const float*)d_in[22], (const float*)d_in[24]};
  float* out = (float*)d_out;

  char* ws = (char*)d_ws;
  size_t off = 0;
  auto carve = [&](size_t bytes) -> void* {
    void* p = ws + off;
    off = (off + bytes + 255) & ~(size_t)255;
    return p;
  };
  _Float16* wIn0  = (_Float16*)carve(96 * 32 * 2);
  _Float16* wIn1  = (_Float16*)carve(96 * 96 * 2);
  _Float16* wIn2  = (_Float16*)carve(96 * 96 * 2);
  _Float16* wIn3  = (_Float16*)carve(96 * 96 * 2);
  _Float16* wMsg0 = (_Float16*)carve(96 * 192 * 2);
  _Float16* wMsg1 = (_Float16*)carve(96 * 96 * 2);
  _Float16* wMsg2 = (_Float16*)carve(96 * 96 * 2);
  _Float16* wMsg3 = (_Float16*)carve(96 * 96 * 2);
  _Float16* wLstm = (_Float16*)carve(336 * 288 * 2);
  _Float16* wOut  = (_Float16*)carve(16 * 96 * 2);
  _Float16* xh    = (_Float16*)carve((size_t)NNODES * HP * 2);
  _Float16* hmsg  = (_Float16*)carve((size_t)NNODES * HP * 2);
  _Float16* hrec  = (_Float16*)carve((size_t)NNODES * HP * 2);
  _Float16* mbuf  = (_Float16*)carve((size_t)NNODES * HP * 2);
  float*    cst   = (float*)   carve((size_t)NNODES * HDIM * 4);
  float*    lossA = (float*)   carve(256);

  // ---- per-launch init (replay safe) ----
  zero_f32_kernel<<<256, 256, 0, stream>>>(cst, NNODES * HDIM);
  zero_f16_kernel<<<256, 256, 0, stream>>>(hrec, NNODES * HP);
  zero_f32_kernel<<<1, 32, 0, stream>>>(lossA, 1);

  // ---- weight conversion to padded f16 ----
  pad_weight_kernel<<<(96 * 32 + 255) / 256, 256, 0, stream>>>(inW[0], wIn0, 84, 16, 96, 32);
  pad_weight_kernel<<<(96 * 96 + 255) / 256, 256, 0, stream>>>(inW[1], wIn1, 84, 84, 96, 96);
  pad_weight_kernel<<<(96 * 96 + 255) / 256, 256, 0, stream>>>(inW[2], wIn2, 84, 84, 96, 96);
  pad_weight_kernel<<<(96 * 96 + 255) / 256, 256, 0, stream>>>(inW[3], wIn3, 84, 84, 96, 96);
  pad_msgw0_kernel<<<(96 * 192 + 255) / 256, 256, 0, stream>>>(msgW[0], wMsg0);
  pad_weight_kernel<<<(96 * 96 + 255) / 256, 256, 0, stream>>>(msgW[1], wMsg1, 84, 84, 96, 96);
  pad_weight_kernel<<<(96 * 96 + 255) / 256, 256, 0, stream>>>(msgW[2], wMsg2, 84, 84, 96, 96);
  pad_weight_kernel<<<(96 * 96 + 255) / 256, 256, 0, stream>>>(msgW[3], wMsg3, 84, 84, 96, 96);
  pack_lstm_kernel<<<(336 * 288 + 255) / 256, 256, 0, stream>>>(lstmWih, lstmWhh, wLstm);
  pad_weight_kernel<<<(16 * 96 + 255) / 256, 256, 0, stream>>>(outW, wOut, 9, 84, 16, 96);

  // ---- input MLP: x = mlp4(emb[q]), hmsg = x ----
  input_mlp_kernel<<<NTILES / 4, 128, 0, stream>>>(q, emb, wIn0, wIn1, wIn2, wIn3,
                                                   inB[0], inB[1], inB[2], inB[3], xh, hmsg);

  // ---- T message-passing + LSTM steps ----
  for (int t = 0; t < TSTEPS; ++t) {
    msg_kernel<<<NTILES, 128, 0, stream>>>(hmsg, src, wMsg0, wMsg1, wMsg2, wMsg3,
                                           msgB[0], msgB[1], msgB[2], msgB[3], mbuf);
    lstm_kernel<<<NTILES, 32, 0, stream>>>(xh, mbuf, hrec, wLstm, cst, hmsg);
    out_kernel<<<NTILES, 32, 0, stream>>>(hrec, wOut, outb, lab, out, lossA, t);
  }
  finalize_kernel<<<1, 1, 0, stream>>>(lossA, out);
}
